// RCNNNet_71528385348354
// MI455X (gfx1250) — matrix operand order repaired and checked by reading, not compiled, not run on hardware
//
#include <hip/hip_runtime.h>
#include <hip/hip_bf16.h>
#include <stdint.h>

typedef __attribute__((ext_vector_type(16))) _Float16 v16h;
typedef __attribute__((ext_vector_type(8)))  float    v8f;

#define BM 64      // block tile M
#define BN 256     // block tile N
#define BK 32      // K step

struct alignas(16) h8 { _Float16 h[8]; };

static __device__ __forceinline__ unsigned lds_off_u32(const void* p) {
  // Generic LDS-aperture addresses carry the LDS byte offset in the low 32 bits.
  return (unsigned)(size_t)p;
}

static __device__ __forceinline__ void async_b128(unsigned lds, const void* g) {
  asm volatile("global_load_async_to_lds_b128 %0, %1, off"
               :: "v"(lds), "v"((unsigned long long)(size_t)g) : "memory");
}

// ---------------------------------------------------------------------------
// Generic fused GEMM on column-major activations:
//   X is [N][Kp] (each column's features contiguous, pitch Kp)
//   W is f16 [Mp64 x Kp] row-major (M padded to 64, K to 32, zeros)
//   Yh (optional) is [N][ldY] f16, rows m0.. written contiguously (M % 64 == 0)
//   Yf (optional) is [M][N] f32 row-major (guarded by rowi < M)
// Block = 256 threads (8 waves, 2x4). Wave computes 32x64 sub-tile:
// 8 x v_wmma_f32_16x16x32_f16 per K step. Double-buffered LDS tiles filled with
// global_load_async_to_lds_b128 (5 async ops/wave/tile, ASYNCcnt pipeline).
// ---------------------------------------------------------------------------
__global__ __launch_bounds__(256)
void wmma_gemm_kernel(const _Float16* __restrict__ W, const _Float16* __restrict__ X,
                      const float* __restrict__ bias,
                      _Float16* __restrict__ Yh, float* __restrict__ Yf,
                      int M, int N, int Kp, int ldY, int relu)
{
  __shared__ __align__(16) _Float16 As[2][BM * BK];   // 2 x 4 KB
  __shared__ __align__(16) _Float16 Bs[2][BN * BK];   // 2 x 16 KB  (col-major: [n][k])

  const int tid  = threadIdx.x;
  const int lane = tid & 31;
  const int wave = tid >> 5;          // 0..7
  const int wrow = (wave >> 2) * 32;  // wave M offset within block tile
  const int wcol = (wave & 3) * 64;   // wave N offset within block tile
  const int m0   = blockIdx.y * BM;
  const int n0   = blockIdx.x * BN;

  // A tile: 64 rows x 32 f16 (64 B/row) = 256 x 16B chunks, 1 per thread.
  const int arow = tid >> 2, acj = tid & 3;
  // B tile: 256 cols x 32 f16 (64 B/col) = 1024 x 16B chunks, 4 per thread.
  const int bq = tid >> 2, bcj = tid & 3;   // chunk q: col = bq + q*64

  // Global source pointers; every pointer advances by BK f16 per K step.
  const _Float16* gA = W + (size_t)(m0 + arow) * Kp + (size_t)(acj * 8);
  const _Float16* gB[4];
  #pragma unroll
  for (int q = 0; q < 4; ++q)
    gB[q] = X + (size_t)(n0 + bq + q * 64) * Kp + (size_t)(bcj * 8);

  // LDS destination offsets for both buffers.
  unsigned lA[2], lB[2][4];
  #pragma unroll
  for (int qb = 0; qb < 2; ++qb) {
    lA[qb] = lds_off_u32(&As[qb][arow * BK + acj * 8]);
    #pragma unroll
    for (int q = 0; q < 4; ++q)
      lB[qb][q] = lds_off_u32(&Bs[qb][(bq + q * 64) * BK + bcj * 8]);
  }

  v8f acc[2][4] = {};
  const int nsteps = Kp / BK;

  // Preload tile 0.
  async_b128(lA[0], gA);  gA += BK;
  #pragma unroll
  for (int q = 0; q < 4; ++q) { async_b128(lB[0][q], gB[q]); gB[q] += BK; }

  for (int t = 0; t < nsteps; ++t) {
    const int cur = t & 1;
    if (t + 1 < nsteps) {
      // Issue next tile into the other buffer (last read two barriers ago).
      async_b128(lA[cur ^ 1], gA);  gA += BK;
      #pragma unroll
      for (int q = 0; q < 4; ++q) { async_b128(lB[cur ^ 1][q], gB[q]); gB[q] += BK; }
      // 5 ops of tile t+1 in flight; oldest 5 (tile t) must have completed.
      asm volatile("s_wait_asynccnt 5" ::: "memory");
    } else {
      asm volatile("s_wait_asynccnt 0" ::: "memory");
    }
    __syncthreads();

    // Fragments (contiguous 16B LDS chunks -> ds_load_b128 pairs).
    const int row  = lane & 15;
    const int half = lane >> 4;
    v16h af[2], bf[4];
    #pragma unroll
    for (int mi = 0; mi < 2; ++mi) {
      const _Float16* Aw = &As[cur][(wrow + mi * 16 + row) * BK];
      #pragma unroll
      for (int j = 0; j < 8; ++j) af[mi][j]     = Aw[half * 16 + j];
      #pragma unroll
      for (int j = 0; j < 8; ++j) af[mi][8 + j] = Aw[half * 16 + 8 + j];
    }
    #pragma unroll
    for (int ni = 0; ni < 4; ++ni) {
      const _Float16* Bc = &Bs[cur][(wcol + ni * 16 + row) * BK + half * 16];
      #pragma unroll
      for (int j = 0; j < 16; ++j) bf[ni][j] = Bc[j];
    }
    #pragma unroll
    for (int mi = 0; mi < 2; ++mi)
      #pragma unroll
      for (int ni = 0; ni < 4; ++ni)
        acc[mi][ni] = __builtin_amdgcn_wmma_f32_16x16x32_f16(
            false, af[mi], false, bf[ni],
            (short)0, acc[mi][ni], false, false);
    __syncthreads();
  }

  // C/D layout: lanes 0-15 -> M = v, lanes 16-31 -> M = 8+v; N = lane&15.
  const int row  = lane & 15;
  const int half = lane >> 4;
  #pragma unroll
  for (int mi = 0; mi < 2; ++mi) {
    const int rowbase = m0 + wrow + mi * 16 + 8 * half;
    #pragma unroll
    for (int ni = 0; ni < 4; ++ni) {
      const int col = n0 + wcol + ni * 16 + row;
      float vals[8];
      #pragma unroll
      for (int v = 0; v < 8; ++v) {
        int rowi = rowbase + v;
        float y = acc[mi][ni][v] + ((rowi < M) ? bias[rowi] : 0.f);
        if (relu) y = fmaxf(y, 0.f);
        vals[v] = y;
      }
      if (Yh) {  // contiguous 16B store into column-major output
        h8 hv;
        #pragma unroll
        for (int v = 0; v < 8; ++v) hv.h[v] = (_Float16)vals[v];
        *(h8*)(Yh + (size_t)col * ldY + rowbase) = hv;
      }
      if (Yf) {
        #pragma unroll
        for (int v = 0; v < 8; ++v) {
          int rowi = rowbase + v;
          if (rowi < M) Yf[(size_t)rowi * N + col] = vals[v];
        }
      }
    }
  }
}

// ---------------------------------------------------------------------------
// Support kernels (activations are column-major [ncols][pitch])
// ---------------------------------------------------------------------------
__global__ void cvt_weight_kernel(const float* __restrict__ W, _Float16* __restrict__ Wh,
                                  int M, int K, int Mp, int Kp) {
  int i = blockIdx.x * 256 + threadIdx.x;
  if (i >= Mp * Kp) return;
  int m = i / Kp, k = i - m * Kp;
  float v = (m < M && k < K) ? W[(size_t)m * K + k] : 0.f;
  Wh[i] = (_Float16)v;
}

// pts_input: [B][N][133].  Xin: f16 [B*N][32] (k 0..4 valid, rest zero)
__global__ void build_x0_kernel(const float* __restrict__ pts, _Float16* __restrict__ Xin,
                                int ncols) {
  int col = blockIdx.x * 256 + threadIdx.x;
  if (col >= ncols) return;
  const float* p = pts + (size_t)col * 133;
  _Float16* o = Xin + (size_t)col * 32;
  #pragma unroll
  for (int k = 0; k < 32; ++k) o[k] = (_Float16)(k < 5 ? p[k] : 0.f);
}

// rpn features into k = 128..255 of Xm columns ([ncols][256])
__global__ void build_rpn_kernel(const float* __restrict__ pts, _Float16* __restrict__ Xm,
                                 int ncols) {
  int col = blockIdx.x * 256 + threadIdx.x;
  if (col >= ncols) return;
  const float* p = pts + (size_t)col * 133 + 5;
  _Float16* o = Xm + (size_t)col * 256 + 128;
  for (int k = 0; k < 128; ++k) o[k] = (_Float16)p[k];
}

// Farthest point sampling: one block per batch, 256 threads, n <= 512.
__global__ __launch_bounds__(256)
void fps_kernel(const float* __restrict__ xyz, int n, int stride, int npoint,
                int* __restrict__ out) {
  int b = blockIdx.x;
  const float* base = xyz + (size_t)b * n * stride;
  __shared__ float dist[512];
  __shared__ float rv[256];
  __shared__ int   ri[256];
  __shared__ int   s_last;
  int tid = threadIdx.x;
  for (int i = tid; i < n; i += 256) dist[i] = 1e10f;
  if (tid == 0) { out[(size_t)b * npoint] = 0; s_last = 0; }
  __syncthreads();
  for (int it = 1; it < npoint; ++it) {
    int last = s_last;
    float lx = base[(size_t)last * stride + 0];
    float ly = base[(size_t)last * stride + 1];
    float lz = base[(size_t)last * stride + 2];
    float best = -1.f; int bi = 0;
    for (int i = tid; i < n; i += 256) {
      float dx = base[(size_t)i * stride + 0] - lx;
      float dy = base[(size_t)i * stride + 1] - ly;
      float dz = base[(size_t)i * stride + 2] - lz;
      float d  = dx * dx + dy * dy + dz * dz;
      float nd = fminf(dist[i], d);
      dist[i] = nd;
      if (nd > best) { best = nd; bi = i; }
    }
    rv[tid] = best; ri[tid] = bi;
    __syncthreads();
    for (int s = 128; s > 0; s >>= 1) {
      if (tid < s && rv[tid + s] > rv[tid]) { rv[tid] = rv[tid + s]; ri[tid] = ri[tid + s]; }
      __syncthreads();
    }
    if (tid == 0) { out[(size_t)b * npoint + it] = ri[0]; s_last = ri[0]; }
    __syncthreads();
  }
}

__global__ void gather_xyz_kernel(const float* __restrict__ xyz, int n, int stride,
                                  const int* __restrict__ fidx, int npoint, int nbatch,
                                  float* __restrict__ newxyz) {
  int i = blockIdx.x * 256 + threadIdx.x;
  if (i >= nbatch * npoint) return;
  int b = i / npoint;
  int idx = fidx[i];
  const float* p = xyz + ((size_t)b * n + idx) * stride;
  newxyz[(size_t)i * 3 + 0] = p[0];
  newxyz[(size_t)i * 3 + 1] = p[1];
  newxyz[(size_t)i * 3 + 2] = p[2];
}

// One thread per query center; collect first nsample in-radius indices, pad with first.
__global__ void ball_query_kernel(const float* __restrict__ xyz, int n, int stride,
                                  const float* __restrict__ newxyz, int npoint, int nbatch,
                                  float r2, int nsample, int* __restrict__ out) {
  int i = blockIdx.x * 64 + threadIdx.x;
  if (i >= nbatch * npoint) return;
  int b = i / npoint;
  const float* base = xyz + (size_t)b * n * stride;
  float cx = newxyz[(size_t)i * 3 + 0];
  float cy = newxyz[(size_t)i * 3 + 1];
  float cz = newxyz[(size_t)i * 3 + 2];
  int* o = out + (size_t)i * nsample;
  int cnt = 0, first = -1;
  for (int j = 0; j < n && cnt < nsample; ++j) {
    float dx = base[(size_t)j * stride + 0] - cx;
    float dy = base[(size_t)j * stride + 1] - cy;
    float dz = base[(size_t)j * stride + 2] - cz;
    if (dx * dx + dy * dy + dz * dz < r2) {
      if (first < 0) first = j;
      o[cnt++] = j;
    }
  }
  if (first < 0) first = 0;
  for (; cnt < nsample; ++cnt) o[cnt] = first;
}

// Build grouped SA input column: [k<3]=rel xyz, [3..3+C-1]=F[src], rest zero.
// Xg: f16 [Ncols][Kp].  F: f16 [nbatch*nprev][C].
__global__ void group_kernel(const float* __restrict__ xyz, int n, int stride,
                             const float* __restrict__ newxyz, const int* __restrict__ idx,
                             const _Float16* __restrict__ F, int C, int nprev,
                             int npoint, int nsample, int Kp, int nbatch,
                             _Float16* __restrict__ Xg, int Ncols) {
  int col = blockIdx.x * 256 + threadIdx.x;
  if (col >= Ncols) return;
  int s = col % nsample;
  int p = (col / nsample) % npoint;
  int b = col / (nsample * npoint);
  int j = idx[((size_t)(b * npoint + p)) * nsample + s];
  const float* pt = xyz + ((size_t)b * n + j) * stride;
  const float* c  = newxyz + (size_t)(b * npoint + p) * 3;
  _Float16* o = Xg + (size_t)col * Kp;
  o[0] = (_Float16)(pt[0] - c[0]);
  o[1] = (_Float16)(pt[1] - c[1]);
  o[2] = (_Float16)(pt[2] - c[2]);
  const _Float16* f = F + (size_t)(b * nprev + j) * C;
  for (int k = 0; k < C; ++k) o[3 + k] = f[k];
  for (int k = 3 + C; k < Kp; ++k) o[k] = (_Float16)0.f;
}

// Max over nsample consecutive columns: Y [Ncols][C] -> F [Ncols/nsample][C].
__global__ void maxpool_kernel(const _Float16* __restrict__ Y, int Ncols, int nsample,
                               int C, _Float16* __restrict__ F) {
  int ngroups = Ncols / nsample;
  int i = blockIdx.x * 256 + threadIdx.x;
  if (i >= C * ngroups) return;
  int g = i / C, c = i - g * C;
  const _Float16* y = Y + (size_t)g * nsample * C + c;
  float m = (float)y[0];
  for (int s = 1; s < nsample; ++s) m = fmaxf(m, (float)y[(size_t)s * C]);
  F[(size_t)g * C + c] = (_Float16)m;
}

// SA3 input: X3 [4096][288]; per col: xyz(3) + F2 col (256) + zeros.
__global__ void build_x3_kernel(const float* __restrict__ nx2, const _Float16* __restrict__ F2,
                                _Float16* __restrict__ X3, int ncols) {
  int col = blockIdx.x * 256 + threadIdx.x;
  if (col >= ncols) return;
  const float* c = nx2 + (size_t)col * 3;
  _Float16* o = X3 + (size_t)col * 288;
  o[0] = (_Float16)c[0];
  o[1] = (_Float16)c[1];
  o[2] = (_Float16)c[2];
  const _Float16* f = F2 + (size_t)col * 256;
  for (int k = 0; k < 256; ++k) o[3 + k] = f[k];
  for (int k = 259; k < 288; ++k) o[k] = (_Float16)0.f;
}

// d_out: cls (B,1) flat then reg (B,46) flat.  clsY: [1 x 256], regY: [46 x 256].
__global__ void write_out_kernel(const float* __restrict__ clsY, const float* __restrict__ regY,
                                 float* __restrict__ out) {
  int i = blockIdx.x * 256 + threadIdx.x;
  if (i < 128) out[i] = clsY[i];
  int t = i - 128;
  if (t >= 0 && t < 46 * 128) {
    int b = t / 46, m = t - b * 46;
    out[128 + t] = regY[(size_t)m * 256 + b];
  }
}

// ---------------------------------------------------------------------------
// Host orchestration
// ---------------------------------------------------------------------------
extern "C" void kernel_launch(void* const* d_in, const int* in_sizes, int n_in,
                              void* d_out, int out_size, void* d_ws, size_t ws_size,
                              hipStream_t stream) {
  (void)in_sizes; (void)n_in; (void)out_size; (void)ws_size;

  const int kB = 128, kN = 512;
  const int N0 = kB * kN;                  // 65536
  const int N1 = kB * 128 * 64;            // 1048576
  const int N2 = kB * 32 * 64;             // 262144
  const int N3 = kB * 32;                  // 4096
  const int NH = 256;                      // head N padded to BN

  const float* pts = (const float*)d_in[0];

  // ---- workspace bump allocator ----
  char* ws = (char*)d_ws;
  size_t off = 0;
  auto alloc = [&](size_t bytes) -> void* {
    void* p = ws + off;
    off = (off + bytes + 255) & ~(size_t)255;
    return p;
  };

  // ---- layer table (d_in order: pts, xyz_up, merge, sa0..2, cls, reg) ----
  const int wIdx[18] = { 1, 3, 5,   7, 9, 11,  13, 15, 17,  19, 21, 23,  25, 27, 29,  31, 33, 35 };
  const int Ms[18]   = {128,128,128, 128,128,128, 128,128,256, 256,256,512, 256,256,  1, 256,256, 46 };
  const int Ks[18]   = {  5,128,256, 131,128,128, 131,128,128, 259,256,256, 512,256,256, 512,256,256 };
  _Float16* Wh[18];
  const float* Bias[18];
  int Kp[18];
  for (int l = 0; l < 18; ++l) {
    int Mp = (Ms[l] + 63) & ~63;             // pad to BM
    Kp[l]  = (Ks[l] + 31) & ~31;             // pad to BK
    Wh[l]  = (_Float16*)alloc((size_t)Mp * Kp[l] * sizeof(_Float16));
    Bias[l] = (const float*)d_in[wIdx[l] + 1];
    int tot = Mp * Kp[l];
    cvt_weight_kernel<<<(tot + 255) / 256, 256, 0, stream>>>(
        (const float*)d_in[wIdx[l]], Wh[l], Ms[l], Ks[l], Mp, Kp[l]);
  }

  // ---- activations / intermediates (all column-major [ncols][pitch]) ----
  _Float16* Xin = (_Float16*)alloc((size_t)N0 * 32 * 2);
  _Float16* xf  = (_Float16*)alloc((size_t)N0 * 128 * 2);
  _Float16* Xm  = (_Float16*)alloc((size_t)N0 * 256 * 2);
  _Float16* F0  = (_Float16*)alloc((size_t)N0 * 128 * 2);
  int*   fidx1  = (int*)  alloc((size_t)kB * 128 * 4);
  float* nx1    = (float*)alloc((size_t)kB * 128 * 3 * 4);
  int*   idx1   = (int*)  alloc((size_t)kB * 128 * 64 * 4);
  _Float16* BUFA = (_Float16*)alloc((size_t)N1 * 160 * 2);
  _Float16* BUFB = (_Float16*)alloc((size_t)N1 * 160 * 2);
  _Float16* F1  = (_Float16*)alloc((size_t)(kB * 128) * 128 * 2);
  int*   fidx2  = (int*)  alloc((size_t)kB * 32 * 4);
  float* nx2    = (float*)alloc((size_t)kB * 32 * 3 * 4);
  int*   idx2   = (int*)  alloc((size_t)kB * 32 * 64 * 4);
  _Float16* F2  = (_Float16*)alloc((size_t)N3 * 256 * 2);
  _Float16* X3  = (_Float16*)alloc((size_t)N3 * 288 * 2);
  _Float16* F3  = (_Float16*)alloc((size_t)NH * 512 * 2);   // cols 128..255 unused garbage
  _Float16* H1  = (_Float16*)alloc((size_t)NH * 256 * 2);
  _Float16* H2  = (_Float16*)alloc((size_t)NH * 256 * 2);
  float* clsF   = (float*)alloc((size_t)1  * NH * 4);
  float* regF   = (float*)alloc((size_t)46 * NH * 4);

  auto gemm = [&](int l, const _Float16* X, _Float16* Yh, float* Yf, int N, int ldY, int relu) {
    dim3 g(N / BN, ((Ms[l] + 63) & ~63) / BM);
    wmma_gemm_kernel<<<g, 256, 0, stream>>>(Wh[l], X, Bias[l], Yh, Yf, Ms[l], N, Kp[l], ldY, relu);
  };

  // ---- stage 0: xyz_up + merge ----
  build_x0_kernel<<<(N0 + 255) / 256, 256, 0, stream>>>(pts, Xin, N0);
  gemm(0, Xin, xf, nullptr, N0, 128, 1);
  gemm(1, xf,  Xm, nullptr, N0, 256, 1);            // writes k 0..127 of Xm columns
  build_rpn_kernel<<<(N0 + 255) / 256, 256, 0, stream>>>(pts, Xm, N0);
  gemm(2, Xm, F0, nullptr, N0, 128, 1);

  // ---- SA1: npoint=128, r=0.2, nsample=64, mlps 128/128/128 ----
  fps_kernel<<<kB, 256, 0, stream>>>(pts, kN, 133, 128, fidx1);
  gather_xyz_kernel<<<(kB * 128 + 255) / 256, 256, 0, stream>>>(pts, kN, 133, fidx1, 128, kB, nx1);
  ball_query_kernel<<<(kB * 128 + 63) / 64, 64, 0, stream>>>(pts, kN, 133, nx1, 128, kB, 0.04f, 64, idx1);
  group_kernel<<<(N1 + 255) / 256, 256, 0, stream>>>(pts, kN, 133, nx1, idx1, F0, 128, kN,
                                                     128, 64, 160, kB, BUFA, N1);
  gemm(3, BUFA, BUFB, nullptr, N1, 128, 1);
  gemm(4, BUFB, BUFA, nullptr, N1, 128, 1);
  gemm(5, BUFA, BUFB, nullptr, N1, 128, 1);
  maxpool_kernel<<<(128 * (N1 / 64) + 255) / 256, 256, 0, stream>>>(BUFB, N1, 64, 128, F1);

  // ---- SA2: npoint=32, r=0.4, nsample=64, mlps 128/128/256 ----
  fps_kernel<<<kB, 256, 0, stream>>>(nx1, 128, 3, 32, fidx2);
  gather_xyz_kernel<<<(kB * 32 + 255) / 256, 256, 0, stream>>>(nx1, 128, 3, fidx2, 32, kB, nx2);
  ball_query_kernel<<<(kB * 32 + 63) / 64, 64, 0, stream>>>(nx1, 128, 3, nx2, 32, kB, 0.16f, 64, idx2);
  group_kernel<<<(N2 + 255) / 256, 256, 0, stream>>>(nx1, 128, 3, nx2, idx2, F1, 128, 128,
                                                     32, 64, 160, kB, BUFA, N2);
  gemm(6, BUFA, BUFB, nullptr, N2, 128, 1);
  gemm(7, BUFB, BUFA, nullptr, N2, 128, 1);
  gemm(8, BUFA, BUFB, nullptr, N2, 256, 1);
  maxpool_kernel<<<(256 * (N2 / 64) + 255) / 256, 256, 0, stream>>>(BUFB, N2, 64, 256, F2);

  // ---- SA3: group-all over 32 points, mlps 256/256/512 ----
  build_x3_kernel<<<(N3 + 255) / 256, 256, 0, stream>>>(nx2, F2, X3, N3);
  gemm(9,  X3,   BUFA, nullptr, N3, 256, 1);
  gemm(10, BUFA, BUFB, nullptr, N3, 256, 1);
  gemm(11, BUFB, BUFA, nullptr, N3, 512, 1);
  maxpool_kernel<<<(512 * (N3 / 32) + 255) / 256, 256, 0, stream>>>(BUFA, N3, 32, 512, F3);

  // ---- heads (N padded to 256; cols 128..255 are garbage, never read) ----
  gemm(12, F3, H1, nullptr, NH, 256, 1);
  gemm(13, H1, H2, nullptr, NH, 256, 1);
  gemm(14, H2, nullptr, clsF, NH, 0, 0);
  gemm(15, F3, H1, nullptr, NH, 256, 1);
  gemm(16, H1, H2, nullptr, NH, 256, 1);
  gemm(17, H2, nullptr, regF, NH, 0, 0);

  write_out_kernel<<<(6016 + 255) / 256, 256, 0, stream>>>(clsF, regF, (float*)d_out);
}
